// StrategyClassifier_39144331935745
// MI455X (gfx1250) — compile-verified, matrix operand order checked by reading, not run on hardware
//
#include <hip/hip_runtime.h>

// Fused GraphSAGE(2-layer) + mean-pool + MLP head for gfx1250 (MI455X).
// One workgroup (128 thr = 4 wave32) processes 16 graphs (= 80 node rows = 5
// WMMA row-tiles). Dense GEMMs run on v_wmma_f32_16x16x32_f16; the 5x5
// adjacency mean-aggregation uses packed-f16 VALU (v_pk_fma_f16) on 16B LDS
// vectors; bias+ReLU epilogues and the mean-pool (LDS ds_add_f32 atomics)
// are fused. Weights are pre-converted to f16 and pre-swizzled into per-lane
// B-fragment order in d_ws (~56 KB of d_ws needed).

typedef __attribute__((ext_vector_type(16))) _Float16 v16h;
typedef __attribute__((ext_vector_type(8)))  _Float16 v8h;
typedef __attribute__((ext_vector_type(8)))  float    v8f;

// d_ws layout (f16 elements)
#define TB_W1   0        //  4 tiles: [W1s;W1n] (32x64)
#define TB_W2   2048     // 32 tiles: [W2s;W2n] (128x128)
#define TB_WC1  18432    // 16 tiles: Wc1 (128x64)
#define TB_WC2  26624    //  2 tiles: Wc2 (64x16, col 15 zero-pad)
#define WS_BIAS_OFF 55296 // bytes; then 272 f32: bias1[64], bias2[128], bc1[64], bc2[16]

// ---------------- weight prep: f32 -> f16, swizzled to B-fragment order ----
// B-fragment (K=32 x N=16 per tile): lane l holds column n=l&15, K half by
// l>>4; VGPR v holds K = (v<4?0:16) + (l>>4)*8 + (v&3)*2 (+odd).  (ISA 7.12.2)
__global__ void prep_weights(const float* __restrict__ W1s, const float* __restrict__ W1n,
                             const float* __restrict__ W2s, const float* __restrict__ W2n,
                             const float* __restrict__ Wc1, const float* __restrict__ Wc2,
                             const float* __restrict__ b1s, const float* __restrict__ b1n,
                             const float* __restrict__ b2s, const float* __restrict__ b2n,
                             const float* __restrict__ bc1, const float* __restrict__ bc2,
                             _Float16* __restrict__ wsW, float* __restrict__ wsB)
{
  int blk = blockIdx.x, lane = threadIdx.x;
  if (blk == 54) {                       // biases
    for (int i = lane; i < 272; i += 32) {
      float v;
      if (i < 64)       v = b1s[i] + b1n[i];
      else if (i < 192) v = b2s[i - 64] + b2n[i - 64];
      else if (i < 256) v = bc1[i - 192];
      else              v = ((i - 256) < 15) ? bc2[i - 256] : 0.f;
      wsB[i] = v;
    }
    return;
  }
  int type, kt, nt, base;
  if (blk < 4)       { type = 0; kt = 0;              nt = blk;           base = TB_W1  + blk * 512; }
  else if (blk < 36) { int t = blk - 4;  type = 1; kt = t >> 3; nt = t & 7; base = TB_W2  + t * 512; }
  else if (blk < 52) { int t = blk - 36; type = 2; kt = t >> 2; nt = t & 3; base = TB_WC1 + t * 512; }
  else               { int t = blk - 52; type = 3; kt = t;      nt = 0;     base = TB_WC2 + t * 512; }
  int nl = lane & 15, hi = lane >> 4;
#pragma unroll
  for (int i = 0; i < 16; ++i) {
    int v = i >> 1, odd = i & 1;
    int kl = ((v < 4) ? 0 : 16) + hi * 8 + (v & 3) * 2 + odd;
    int k = kt * 32 + kl;
    int n = nt * 16 + nl;
    float val;
    if (type == 0)      val = (k < 16) ? W1s[k * 64 + n]  : W1n[(k - 16) * 64 + n];
    else if (type == 1) val = (k < 64) ? W2s[k * 128 + n] : W2n[(k - 64) * 128 + n];
    else if (type == 2) val = Wc1[k * 64 + n];
    else                val = (n < 15) ? Wc2[k * 15 + n] : 0.f;
    wsW[base + lane * 16 + i] = (_Float16)val;
  }
}

// A-fragment gather from LDS (16-bit A 16x32 layout, ISA 7.12.2)
__device__ inline v16h load_a_frag(const _Float16* base, int stride, int row0, int kbase) {
  int lane = threadIdx.x & 31;
  int m = lane & 15, hi = lane >> 4;
  const _Float16* rp = base + (row0 + m) * stride + kbase + hi * 8;
  v16h a;
#pragma unroll
  for (int v = 0; v < 8; ++v) {
    int kl = ((v < 4) ? 0 : 16) + (v & 3) * 2;   // even -> dword-aligned pair
    a[2 * v]     = rp[kl];
    a[2 * v + 1] = rp[kl + 1];
  }
  return a;
}

__device__ inline v16h load_b_frag(const _Float16* tile) {
  int lane = threadIdx.x & 31;
  return ((const v16h*)tile)[lane];               // 32B contiguous per lane
}

__device__ inline v8f wmma_f16(v16h a, v16h b, v8f c) {
  return __builtin_amdgcn_wmma_f32_16x16x32_f16(false, a, false, b, (short)0, c, false, false);
}

#define GPW 16  // graphs per workgroup (80 rows = 5 row-tiles)

__global__ __launch_bounds__(128) void sage_fused(
    const float* __restrict__ x, const float* __restrict__ adj,
    const _Float16* __restrict__ wsW, const float* __restrict__ wsB,
    float* __restrict__ out, int Btot)
{
  __shared__ _Float16 wadjh[80][8];      // adj row / clamped degree (f16)
  __shared__ _Float16 A1[80][40];        // [x | neigh_x]   K=32 (+pad, rows 16B-aligned)
  __shared__ _Float16 H1[80][136];       // [h1 | neigh_h1] K=128 (+pad, rows 16B-aligned)
  __shared__ float    pooled[16][128];   // mean-pool accumulators (f32)
  __shared__ _Float16 Pf[16][136];       // pooled as f16 A-matrix
  __shared__ _Float16 Hid[16][72];       // classifier hidden (f16)

  const int tid  = threadIdx.x;
  const int lane = tid & 31;
  const int wave = tid >> 5;
  const long g0  = (long)blockIdx.x * GPW;

  // zero pool accumulators
  for (int idx = tid; idx < 16 * 128; idx += 128)
    (&pooled[0][0])[idx] = 0.f;

  // ---- Phase 0/1a: degrees + scaled adjacency (f16), x -> f16 into A1[:,0:16]
  if (tid < 80) {
    int g = tid / 5, i = tid % 5;
    long gg = g0 + g;
    if (gg < Btot) {
      const float* ar = adj + gg * 25 + i * 5;
      float a[5], s = 0.f;
#pragma unroll
      for (int j = 0; j < 5; ++j) { a[j] = ar[j]; s += a[j]; }
      float inv = 1.f / fmaxf(s, 1.f);
#pragma unroll
      for (int j = 0; j < 5; ++j) wadjh[tid][j] = (_Float16)(a[j] * inv);
      const float4* xr = (const float4*)(x + (gg * 5 + i) * 16);
#pragma unroll
      for (int c = 0; c < 2; ++c) {                 // two 8-wide f16 chunks
        float4 f0 = xr[2 * c], f1 = xr[2 * c + 1];
        v8h h = { (_Float16)f0.x, (_Float16)f0.y, (_Float16)f0.z, (_Float16)f0.w,
                  (_Float16)f1.x, (_Float16)f1.y, (_Float16)f1.z, (_Float16)f1.w };
        *(v8h*)&A1[tid][c * 8] = h;                 // ds_store_b128
      }
    } else {
#pragma unroll
      for (int j = 0; j < 5; ++j) wadjh[tid][j] = (_Float16)0.f;
      v8h z = {};
      *(v8h*)&A1[tid][0] = z;
      *(v8h*)&A1[tid][8] = z;
    }
  }
  __syncthreads();

  // ---- Phase 1b: neigh_x = (adj/deg) @ x -> A1[:,16:32]   (packed f16 FMA)
  if (tid < 80) {
    int g = tid / 5;
    _Float16 w[5];
#pragma unroll
    for (int j = 0; j < 5; ++j) w[j] = wadjh[tid][j];
#pragma unroll
    for (int c = 0; c < 2; ++c) {
      v8h acc = {};
#pragma unroll
      for (int j = 0; j < 5; ++j)
        acc += (*(const v8h*)&A1[g * 5 + j][c * 8]) * w[j];   // v_pk_fma_f16
      *(v8h*)&A1[tid][16 + c * 8] = acc;
    }
  }
  __syncthreads();

  // ---- Phase 2: L1 GEMM  [x|neigh](80x32) @ [W1s;W1n](32x64), bias+ReLU
  for (int t = wave; t < 20; t += 4) {         // nt constant per wave
    int rt = t >> 2, nt = t & 3;
    v16h a = load_a_frag(&A1[0][0], 40, rt * 16, 0);
    v16h b = load_b_frag(wsW + TB_W1 + nt * 512);
    v8f c = {};
    c = wmma_f16(a, b, c);
    int n = lane & 15, hi = lane >> 4;
    float bia = wsB[nt * 16 + n];
#pragma unroll
    for (int j = 0; j < 8; ++j) {
      int row = rt * 16 + j + 8 * hi;
      H1[row][nt * 16 + n] = (_Float16)fmaxf(c[j] + bia, 0.f);
    }
  }
  __syncthreads();

  // ---- Phase 3: neigh_h1 = (adj/deg) @ h1 -> H1[:,64:128]  (packed f16 FMA)
  for (int task = tid; task < 160; task += 128) {
    int row = task >> 1, half = task & 1, g = row / 5;
    int k0 = half * 32;
    _Float16 w[5];
#pragma unroll
    for (int j = 0; j < 5; ++j) w[j] = wadjh[row][j];
    v8h acc[4] = {};
#pragma unroll
    for (int j = 0; j < 5; ++j) {
      const _Float16* src = &H1[g * 5 + j][k0];
#pragma unroll
      for (int c = 0; c < 4; ++c)
        acc[c] += (*(const v8h*)(src + c * 8)) * w[j];        // ds_load_b128 + v_pk_fma_f16
    }
    _Float16* dst = &H1[row][64 + k0];
#pragma unroll
    for (int c = 0; c < 4; ++c)
      *(v8h*)(dst + c * 8) = acc[c];                          // ds_store_b128
  }
  __syncthreads();

  // ---- Phase 4: L2 GEMM  [h1|neigh](80x128) @ [W2s;W2n](128x128),
  //      bias+ReLU+mean-pool fused (ds_add_f32 atomics, x 1/5)
  const float* bias2 = wsB + 64;
  for (int t = wave; t < 40; t += 4) {
    int rt = t >> 3, nt = t & 7;
    v8f c = {};
#pragma unroll
    for (int kt = 0; kt < 4; ++kt) {
      v16h a = load_a_frag(&H1[0][0], 136, rt * 16, kt * 32);
      v16h b = load_b_frag(wsW + TB_W2 + (kt * 8 + nt) * 512);
      c = wmma_f16(a, b, c);
    }
    int n = lane & 15, hi = lane >> 4;
    int nn = nt * 16 + n;
    float bia = bias2[nn];
#pragma unroll
    for (int j = 0; j < 8; ++j) {
      int row = rt * 16 + j + 8 * hi;
      int g = row / 5;
      float vv = fmaxf(c[j] + bia, 0.f);
      atomicAdd(&pooled[g][nn], vv * 0.2f);
    }
  }
  __syncthreads();

  // pooled f32 -> f16 A-matrix
  for (int idx = tid; idx < 16 * 128; idx += 128) {
    int g = idx >> 7, n = idx & 127;
    Pf[g][n] = (_Float16)pooled[g][n];
  }
  __syncthreads();

  // ---- Phase 5: head L1: pooled(16x128) @ Wc1(128x64), bias+ReLU
  {
    int nt = wave;                                // 4 waves -> 4 n-tiles
    v8f c = {};
#pragma unroll
    for (int kt = 0; kt < 4; ++kt) {
      v16h a = load_a_frag(&Pf[0][0], 136, 0, kt * 32);
      v16h b = load_b_frag(wsW + TB_WC1 + (kt * 4 + nt) * 512);
      c = wmma_f16(a, b, c);
    }
    int n = lane & 15, hi = lane >> 4;
    float bia = wsB[192 + nt * 16 + n];
#pragma unroll
    for (int j = 0; j < 8; ++j) {
      int g = j + 8 * hi;
      Hid[g][nt * 16 + n] = (_Float16)fmaxf(c[j] + bia, 0.f);
    }
  }
  __syncthreads();

  // ---- Phase 6: head L2: hid(16x64) @ Wc2(64x16pad) -> logits (cols 0..14)
  if (wave == 0) {
    v8f c = {};
#pragma unroll
    for (int kt = 0; kt < 2; ++kt) {
      v16h a = load_a_frag(&Hid[0][0], 72, 0, kt * 32);
      v16h b = load_b_frag(wsW + TB_WC2 + kt * 512);
      c = wmma_f16(a, b, c);
    }
    int n = lane & 15, hi = lane >> 4;
    if (n < 15) {                                 // divergence only at store
      float bia = wsB[256 + n];
#pragma unroll
      for (int j = 0; j < 8; ++j) {
        long gg = g0 + j + 8 * hi;
        if (gg < Btot) out[gg * 15 + n] = c[j] + bia;
      }
    }
  }
}

extern "C" void kernel_launch(void* const* d_in, const int* in_sizes, int n_in,
                              void* d_out, int out_size, void* d_ws, size_t ws_size,
                              hipStream_t stream) {
  const float* x   = (const float*)d_in[0];
  const float* adj = (const float*)d_in[1];
  const float* W1s = (const float*)d_in[2];
  const float* b1s = (const float*)d_in[3];
  const float* W1n = (const float*)d_in[4];
  const float* b1n = (const float*)d_in[5];
  const float* W2s = (const float*)d_in[6];
  const float* b2s = (const float*)d_in[7];
  const float* W2n = (const float*)d_in[8];
  const float* b2n = (const float*)d_in[9];
  const float* Wc1 = (const float*)d_in[10];
  const float* bc1 = (const float*)d_in[11];
  const float* Wc2 = (const float*)d_in[12];
  const float* bc2 = (const float*)d_in[13];
  float* out = (float*)d_out;

  _Float16* wsW = (_Float16*)d_ws;
  float*    wsB = (float*)((char*)d_ws + WS_BIAS_OFF);

  int Btot = in_sizes[0] / 80;              // B*5*16 -> B

  prep_weights<<<55, 32, 0, stream>>>(W1s, W1n, W2s, W2n, Wc1, Wc2,
                                      b1s, b1n, b2s, b2n, bc1, bc2, wsW, wsB);

  int grid = (Btot + GPW - 1) / GPW;        // 16384 for B=262144
  sage_fused<<<grid, 128, 0, stream>>>(x, adj, wsW, wsB, out, Btot);
}
